// SplitCNN_56014963474544
// MI455X (gfx1250) — matrix operand order repaired and checked by reading, not compiled, not run on hardware
//
#include <hip/hip_runtime.h>
#include <math.h>

// ---------------------------------------------------------------------------
// Types for WMMA / TDM (gfx1250, wave32)
// ---------------------------------------------------------------------------
typedef __attribute__((ext_vector_type(16))) __bf16 v16bf;
typedef __attribute__((ext_vector_type(8)))  float  v8f;
typedef __attribute__((ext_vector_type(4)))  unsigned int v4u;
typedef __attribute__((ext_vector_type(8)))  int    v8i;
typedef __attribute__((ext_vector_type(4)))  int    v4i;

union Frag16 {
    v16bf v;
    v4u   u[2];
};

#if defined(__has_builtin)
#if __has_builtin(__builtin_amdgcn_tensor_load_to_lds) && \
    __has_builtin(__builtin_amdgcn_s_wait_tensorcnt)
#define USE_TDM 1
#endif
#endif
#ifndef USE_TDM
#define USE_TDM 0
#endif

// Sizes
#define BATCH 4096
#define NDEV  10
#define JDIM  256

// GEMM tiling
#define KC        32            // K per chunk (one bf16 WMMA step)
#define NCHUNK    128           // total K chunks: 2*(1024/32) + 8*(256/32)
#define BROWS     64            // j rows staged per chunk (one N quarter)
#define RS        80            // LDS row stride bytes (64B data + 16B pad)
#define BUF_BYTES (BROWS * RS)  // 5120 B per buffer

// Workspace layout (bytes)
#define FLAT_ELEMS   (2ull*BATCH*1024ull + 8ull*BATCH*256ull)   // 16,777,216 bf16
#define WBF_ELEMS    (2ull*256*1024ull + 8ull*256*256ull)       //  1,048,576 bf16
#define OFF_FLAT     0ull
#define OFF_WBF      (OFF_FLAT + FLAT_ELEMS*2ull)
#define OFF_RECV     (OFF_WBF  + WBF_ELEMS*2ull)
#define OFF_COEF     (OFF_RECV + (size_t)BATCH*JDIM*4ull)
#define OFF_NOISE    (OFF_COEF + (size_t)NDEV*JDIM*4ull)

__device__ __forceinline__ size_t flat_off(int n) {
    return (n < 2) ? (size_t)n * BATCH * 1024ull
                   : 2ull * BATCH * 1024ull + (size_t)(n - 2) * BATCH * 256ull;
}
__device__ __forceinline__ size_t wbf_off(int n) {
    return (n < 2) ? (size_t)n * 256ull * 1024ull
                   : 2ull * 256ull * 1024ull + (size_t)(n - 2) * 256ull * 256ull;
}
// chunk c -> (device n, k offset, device K)
__device__ __forceinline__ void chunk_map(int c, int& n, int& k0, int& Kn) {
    if (c < 64) { n = c >> 5;            k0 = (c & 31) * KC; Kn = 1024; }
    else        { int cc = c - 64;
                  n = 2 + (cc >> 3);     k0 = (cc & 7) * KC; Kn = 256;  }
}

// ---------------------------------------------------------------------------
// Kernel 1: coefficient / noise precompute.
//   A[j,c]     = sum_m a_mat[j,c,m]
//   coef[n,j]  = b_mat[n,j] * sum_c (sum_k ind[j,k]*h_mat[n,k,c]) * A[j,c]
//   noise_t[j] = sum_c noise[j,c]*A[j,c]
// ---------------------------------------------------------------------------
__global__ __launch_bounds__(256) void prep_coef_kernel(
    const float* __restrict__ b_mat,     // (10,256)
    const float* __restrict__ h_mat,     // (10,64,5)
    const float* __restrict__ a_mat,     // (256,5,5)
    const float* __restrict__ indicator, // (256,64)
    const float* __restrict__ noise,     // (256,5)
    float* __restrict__ coef,            // (10,256)
    float* __restrict__ noise_t)         // (256)
{
    int j = threadIdx.x;
    float A[5];
#pragma unroll
    for (int c = 0; c < 5; ++c) {
        float s = 0.f;
#pragma unroll
        for (int m = 0; m < 5; ++m) s += a_mat[(j*5 + c)*5 + m];
        A[c] = s;
    }
    float nt = 0.f;
#pragma unroll
    for (int c = 0; c < 5; ++c) nt += noise[j*5 + c] * A[c];
    noise_t[j] = nt;

    for (int n = 0; n < NDEV; ++n) {
        float hv = 0.f;
#pragma unroll
        for (int c = 0; c < 5; ++c) {
            float hs = 0.f;
            for (int k = 0; k < 64; ++k)
                hs += indicator[j*64 + k] * h_mat[(n*64 + k)*5 + c];
            hv += hs * A[c];
        }
        coef[n*JDIM + j] = b_mat[n*JDIM + j] * hv;
    }
}

// ---------------------------------------------------------------------------
// Kernel 2: cut-weights fp32 -> coef-scaled bf16 (2 big then 8 small, row-major)
// ---------------------------------------------------------------------------
__global__ __launch_bounds__(256) void prep_wbf_kernel(
    const float* __restrict__ w_big,   // (2,256,1024)
    const float* __restrict__ w_small, // (8,256,256)
    const float* __restrict__ coef,    // (10,256)
    __bf16* __restrict__ wbf)
{
    size_t i = (size_t)blockIdx.x * blockDim.x + threadIdx.x;
    if (i >= WBF_ELEMS) return;
    int n, j;
    float v;
    if (i < 524288ull) {
        n = (int)(i >> 18);            // / (256*1024)
        j = (int)((i >> 10) & 255);
        v = w_big[i];
    } else {
        size_t ii = i - 524288ull;
        n = 2 + (int)(ii >> 16);       // / (256*256)
        j = (int)((ii >> 8) & 255);
        v = w_small[ii];
    }
    wbf[i] = (__bf16)(v * coef[n*JDIM + j]);
}

// ---------------------------------------------------------------------------
// Kernel 3: per-device branch: crop -> conv3x3(pad1) -> bias+BN -> ReLU ->
//           maxpool2x2 -> flatten -> bf16.  grid = (4096, 10); block = 256
// ---------------------------------------------------------------------------
__global__ __launch_bounds__(256) void conv_branch_kernel(
    const float* __restrict__ x,       // (B,3,64,64)
    const float* __restrict__ conv_w,  // (10,4,3,3,3)
    const float* __restrict__ conv_b,  // (10,4)
    const float* __restrict__ bn_g,    // (10,4)
    const float* __restrict__ bn_b,    // (10,4)
    const float* __restrict__ bn_m,    // (10,4)
    const float* __restrict__ bn_v,    // (10,4)
    __bf16* __restrict__ flat)
{
    __shared__ float tile[3*34*34];
    __shared__ float wsm[108];
    __shared__ float scl[4], sft[4];

    const int b   = blockIdx.x;
    const int n   = blockIdx.y;
    const int tid = threadIdx.x;

    int S, R0, C0;
    if (n <= 1) { S = 32; R0 = 32*n; C0 = 0; }
    else {
        int m = n - 2;
        S = 16; R0 = 16*(m & 3); C0 = 32 + 16*(m >> 2);
    }
    const int T = S + 2;
    const int tot_pad = 3*T*T;

    for (int i = tid; i < tot_pad; i += 256) tile[i] = 0.f;
    if (tid < 108) wsm[tid] = conv_w[n*108 + tid];
    if (tid < 4) {
        float s = bn_g[n*4 + tid] * rsqrtf(bn_v[n*4 + tid] + 1e-5f);
        scl[tid] = s;
        sft[tid] = (conv_b[n*4 + tid] - bn_m[n*4 + tid]) * s + bn_b[n*4 + tid];
    }
    __syncthreads();

    const int tot_in = 3*S*S;
    for (int i = tid; i < tot_in; i += 256) {
        int ci = i / (S*S);
        int rm = i - ci*S*S;
        int r  = rm / S, c = rm - r*S;
        tile[ci*T*T + (r+1)*T + (c+1)] =
            x[(((size_t)b*3 + ci)*64 + (R0 + r))*64 + (C0 + c)];
    }
    __syncthreads();

    const int P  = S >> 1;
    const int Kn = 4*P*P;
    const size_t base = flat_off(n) + (size_t)b * Kn;

    for (int f = tid; f < Kn; f += 256) {
        int c  = f / (P*P);
        int rm = f - c*P*P;
        int pi = rm / P, pj = rm - pi*P;
        float mx = 0.f;
#pragma unroll
        for (int dy = 0; dy < 2; ++dy) {
#pragma unroll
            for (int dx = 0; dx < 2; ++dx) {
                int y = 2*pi + dy, xx = 2*pj + dx;
                float acc = 0.f;
#pragma unroll
                for (int ci = 0; ci < 3; ++ci)
#pragma unroll
                    for (int ky = 0; ky < 3; ++ky)
#pragma unroll
                        for (int kx = 0; kx < 3; ++kx)
                            acc += tile[ci*T*T + (y+ky)*T + (xx+kx)] *
                                   wsm[((c*3 + ci)*3 + ky)*3 + kx];
                float v = fmaxf(acc * scl[c] + sft[c], 0.f);
                mx = fmaxf(mx, v);
            }
        }
        flat[base + f] = (__bf16)mx;
    }
}

// ---------------------------------------------------------------------------
// A-fragment load (16-bit A 16x32, contiguous k, same order as B):
// lanes 0-15 = rows, K 0..15; lanes 16-31 = K 16..31
// ---------------------------------------------------------------------------
__device__ __forceinline__ v16bf load_a_frag(const __bf16* base, int stride, int lane)
{
    int r  = lane & 15;
    int kg = (lane >> 4) << 4;
    const __bf16* p = base + (size_t)r * stride + kg;
    Frag16 f;
    f.u[0] = *reinterpret_cast<const v4u*>(p);
    f.u[1] = *reinterpret_cast<const v4u*>(p + 8);
    return f.v;
}

// ---------------------------------------------------------------------------
// TDM: stage 64 weight rows x 32 k (bf16) into LDS, 80B row stride.
// D# per ISA 8.3/8.4: 2D tensor, data_size=2B, pad 4 DW per 16-DW interval.
// 6-arg builtin (clang-23 / therock-10.0): groups 2/3 + extra group unused
// for 2D tensors -> zero-filled.
// ---------------------------------------------------------------------------
#if USE_TDM
__device__ __forceinline__ void tdm_issue_b(const __bf16* gWeights, int Kn,
                                            int jq, int k0, unsigned lds_addr)
{
    unsigned long long ga = (unsigned long long)(uintptr_t)gWeights
                          + ((unsigned long long)jq * (unsigned)Kn + (unsigned)k0) * 2ull;
    v4u g0;
    g0.x = 1u;                                       // count=1, user mode
    g0.y = lds_addr;                                 // LDS byte address
    g0.z = (unsigned)(ga & 0xFFFFFFFFull);           // global_addr[31:0]
    g0.w = (unsigned)((ga >> 32) & 0x01FFFFFFull)    // global_addr[56:32]
         | (2u << 30);                               // type=2 (image)
    v8i g1;
    g1[0] = (int)((1u << 16)      // data_size = 1 -> 2 bytes
                | (1u << 20)      // pad_enable
                | (3u << 22)      // pad_interval code 3 -> every 16 DW (64B)
                | (3u << 25));    // pad_amount  code 3 -> 4 DW (16B)
    g1[1] = (int)(((unsigned)Kn & 0xFFFFu) << 16);            // tensor_dim0 lo16
    g1[2] = (int)((((unsigned)Kn >> 16) & 0xFFFFu)            // tensor_dim0 hi16
                | ((unsigned)BROWS << 16));                   // tensor_dim1 lo16
    g1[3] = (int)((unsigned)KC << 16);                        // dim1 hi16=0, tile_dim0=32
    g1[4] = (int)((unsigned)BROWS);                           // tile_dim1=64, tile_dim2=0
    g1[5] = (int)Kn;                                          // tensor_dim0_stride lo32
    g1[6] = 0;                                                // stride hi, dim1_stride lo
    g1[7] = 0;
    v4i z4 = {0, 0, 0, 0};
    v8i z8 = {0, 0, 0, 0, 0, 0, 0, 0};
    __builtin_amdgcn_tensor_load_to_lds(g0, g1, z4, z4, z8, 0);
}
#endif

// ---------------------------------------------------------------------------
// Kernel 4: fused K=4096 cut-GEMM (coef pre-folded into weights) + noise + ReLU
// grid = (4096/64, 4) ; block = 128 (4 waves x 16 rows -> M=64 per block);
// each block owns a 64-wide N quarter. B tiles staged in LDS (TDM double-
// buffered, 1 barrier per chunk), A fragments from global b128 loads.
// ---------------------------------------------------------------------------
__global__ __launch_bounds__(128) void gemm_combine_kernel(
    const __bf16* __restrict__ flat,
    const __bf16* __restrict__ wsc,     // coef-scaled bf16 weights
    const float*  __restrict__ noise_t,
    float* __restrict__ recv)
{
    __shared__ __align__(16) unsigned char ldsB[2 * BUF_BYTES];

    const int lane = threadIdx.x & 31;
    const int wave = threadIdx.x >> 5;
    const int mrow = blockIdx.x * 64 + wave * 16;   // this wave's 16 batch rows
    const int jq   = blockIdx.y * 64;               // N quarter

    v8f acc[4] = {v8f(0.f), v8f(0.f), v8f(0.f), v8f(0.f)};

#if USE_TDM
    const unsigned lds0 = (unsigned)(uintptr_t)(&ldsB[0]);  // LDS aperture low 32
    if (wave == 0) {
        int n, k0, Kn; chunk_map(0, n, k0, Kn);
        tdm_issue_b(wsc + wbf_off(n), Kn, jq, k0, lds0);
    }
    for (int c = 0; c < NCHUNK; ++c) {
        const int cur = c & 1;
        if (wave == 0) __builtin_amdgcn_s_wait_tensorcnt(0);   // chunk c landed
        __syncthreads();                                        // visible to all
        if (wave == 0 && c + 1 < NCHUNK) {                      // prefetch c+1
            int n, k0, Kn; chunk_map(c + 1, n, k0, Kn);
            tdm_issue_b(wsc + wbf_off(n), Kn, jq, k0,
                        lds0 + (unsigned)((cur ^ 1) * BUF_BYTES));
        }
        int n, k0, Kn; chunk_map(c, n, k0, Kn);
        const __bf16* aB = flat + flat_off(n) + (size_t)mrow * Kn + k0;
        v16bf a = load_a_frag(aB, Kn, lane);
        const unsigned char* bbuf = &ldsB[cur * BUF_BYTES];
#pragma unroll
        for (int t = 0; t < 4; ++t) {
            // B 32x16 (contiguous k, as A): lanes 0-15 = cols (K 0..15), lanes 16-31 = K 16..31
            const v4u* p = reinterpret_cast<const v4u*>(
                bbuf + (t * 16 + (lane & 15)) * RS + ((lane >> 4) * 32));
            Frag16 f; f.u[0] = p[0]; f.u[1] = p[1];
            acc[t] = __builtin_amdgcn_wmma_f32_16x16x32_bf16(
                false, a, false, f.v, (short)0, acc[t], false, false);
        }
    }
#else
    for (int c = 0; c < NCHUNK; ++c) {
        const int cur = c & 1;
        int n, k0, Kn; chunk_map(c, n, k0, Kn);
        __syncthreads();
        {   // cooperative stage: 128 threads x 32B = 64 rows x 64B
            int row = threadIdx.x >> 1, half = threadIdx.x & 1;
            const v4u* src = reinterpret_cast<const v4u*>(
                wsc + wbf_off(n) + (size_t)(jq + row) * Kn + k0 + half * 16);
            v4u* dst = reinterpret_cast<v4u*>(
                &ldsB[cur * BUF_BYTES] + row * RS + half * 32);
            dst[0] = src[0]; dst[1] = src[1];
        }
        __syncthreads();
        const __bf16* aB = flat + flat_off(n) + (size_t)mrow * Kn + k0;
        v16bf a = load_a_frag(aB, Kn, lane);
        const unsigned char* bbuf = &ldsB[cur * BUF_BYTES];
#pragma unroll
        for (int t = 0; t < 4; ++t) {
            const v4u* p = reinterpret_cast<const v4u*>(
                bbuf + (t * 16 + (lane & 15)) * RS + ((lane >> 4) * 32));
            Frag16 f; f.u[0] = p[0]; f.u[1] = p[1];
            acc[t] = __builtin_amdgcn_wmma_f32_16x16x32_bf16(
                false, a, false, f.v, (short)0, acc[t], false, false);
        }
    }
#endif

    // epilogue: + noise, ReLU, store f32 (C/D layout: lanes 16-31 = rows 8..15)
    const int col0   = jq + (lane & 15);
    const int rowadd = (lane >> 4) * 8;
#pragma unroll
    for (int t = 0; t < 4; ++t) {
        float nz = noise_t[col0 + t*16];
#pragma unroll
        for (int r = 0; r < 8; ++r) {
            float v = fmaxf(acc[t][r] + nz, 0.f);
            recv[(size_t)(mrow + rowadd + r) * JDIM + col0 + t*16] = v;
        }
    }
}

// ---------------------------------------------------------------------------
// Kernel 5: MLP head 256->128->64->10 + log_softmax.
// grid = 256 blocks x 256 threads; each block handles 16 batch rows.
// ---------------------------------------------------------------------------
__global__ __launch_bounds__(256) void mlp_head_kernel(
    const float* __restrict__ recv,    // (B,256) post-ReLU
    const float* __restrict__ fc1_w,   // (128,256)
    const float* __restrict__ fc1_b,   // (128)
    const float* __restrict__ fc2_w,   // (64,128)
    const float* __restrict__ fc2_b,   // (64)
    const float* __restrict__ out_w,   // (10,64)
    const float* __restrict__ out_b,   // (10)
    float* __restrict__ out)           // (B,10) log-probs
{
    __shared__ float h1s[16*256];
    __shared__ float h2s[16*128];
    __shared__ float h3s[16*64];
    __shared__ float lg[16*10];

    const int tid   = threadIdx.x;
    const int rbase = blockIdx.x * 16;

    for (int i = tid; i < 16*256; i += 256)
        h1s[i] = recv[(size_t)rbase * 256 + i];
    __syncthreads();

    {   // fc1: o = tid&127, row-half = tid>>7
        int o  = tid & 127;
        int r0 = (tid >> 7) * 8;
        float acc[8] = {0,0,0,0,0,0,0,0};
        for (int k = 0; k < 256; ++k) {
            float w = fc1_w[o*256 + k];
#pragma unroll
            for (int rr = 0; rr < 8; ++rr) acc[rr] += h1s[(r0+rr)*256 + k] * w;
        }
        float b = fc1_b[o];
#pragma unroll
        for (int rr = 0; rr < 8; ++rr)
            h2s[(r0+rr)*128 + o] = fmaxf(acc[rr] + b, 0.f);
    }
    __syncthreads();

    {   // fc2: o = tid&63, row-quarter = tid>>6
        int o  = tid & 63;
        int r0 = (tid >> 6) * 4;
        float acc[4] = {0,0,0,0};
        for (int k = 0; k < 128; ++k) {
            float w = fc2_w[o*128 + k];
#pragma unroll
            for (int rr = 0; rr < 4; ++rr) acc[rr] += h2s[(r0+rr)*128 + k] * w;
        }
        float b = fc2_b[o];
#pragma unroll
        for (int rr = 0; rr < 4; ++rr)
            h3s[(r0+rr)*64 + o] = fmaxf(acc[rr] + b, 0.f);
    }
    __syncthreads();

    if (tid < 160) {   // out: 10 logits x 16 rows
        int r = tid / 10, o = tid - r*10;
        float acc = 0.f;
#pragma unroll
        for (int k = 0; k < 64; ++k) acc += h3s[r*64 + k] * out_w[o*64 + k];
        lg[r*10 + o] = acc + out_b[o];
    }
    __syncthreads();

    if (tid < 16) {    // log_softmax per row
        int r = tid;
        float mx = lg[r*10];
#pragma unroll
        for (int o = 1; o < 10; ++o) mx = fmaxf(mx, lg[r*10 + o]);
        float s = 0.f;
#pragma unroll
        for (int o = 0; o < 10; ++o) s += expf(lg[r*10 + o] - mx);
        float lse = mx + logf(s);
#pragma unroll
        for (int o = 0; o < 10; ++o)
            out[(size_t)(rbase + r)*10 + o] = lg[r*10 + o] - lse;
    }
}

// ---------------------------------------------------------------------------
// Launcher
// ---------------------------------------------------------------------------
extern "C" void kernel_launch(void* const* d_in, const int* in_sizes, int n_in,
                              void* d_out, int out_size, void* d_ws, size_t ws_size,
                              hipStream_t stream) {
    (void)in_sizes; (void)n_in; (void)out_size; (void)ws_size;

    const float* x         = (const float*)d_in[0];
    const float* conv_w    = (const float*)d_in[1];
    const float* conv_b    = (const float*)d_in[2];
    const float* bn_gamma  = (const float*)d_in[3];
    const float* bn_beta   = (const float*)d_in[4];
    const float* bn_mean   = (const float*)d_in[5];
    const float* bn_var    = (const float*)d_in[6];
    const float* cut_big   = (const float*)d_in[7];
    const float* cut_small = (const float*)d_in[8];
    const float* fc1_w     = (const float*)d_in[9];
    const float* fc1_b     = (const float*)d_in[10];
    const float* fc2_w     = (const float*)d_in[11];
    const float* fc2_b     = (const float*)d_in[12];
    const float* out_w     = (const float*)d_in[13];
    const float* out_b     = (const float*)d_in[14];
    const float* b_mat     = (const float*)d_in[15];
    const float* h_mat     = (const float*)d_in[16];
    const float* a_mat     = (const float*)d_in[17];
    const float* indicator = (const float*)d_in[18];
    const float* noise     = (const float*)d_in[19];

    char*   ws      = (char*)d_ws;
    __bf16* flat    = (__bf16*)(ws + OFF_FLAT);
    __bf16* wbf     = (__bf16*)(ws + OFF_WBF);
    float*  recv    = (float*)(ws + OFF_RECV);
    float*  coef    = (float*)(ws + OFF_COEF);
    float*  noise_t = (float*)(ws + OFF_NOISE);

    prep_coef_kernel<<<1, 256, 0, stream>>>(b_mat, h_mat, a_mat, indicator, noise,
                                            coef, noise_t);
    prep_wbf_kernel<<<(unsigned)((WBF_ELEMS + 255) / 256), 256, 0, stream>>>(
        cut_big, cut_small, coef, wbf);
    conv_branch_kernel<<<dim3(BATCH, NDEV), 256, 0, stream>>>(
        x, conv_w, conv_b, bn_gamma, bn_beta, bn_mean, bn_var, flat);
    gemm_combine_kernel<<<dim3(BATCH / 64, 4), 128, 0, stream>>>(
        flat, wbf, noise_t, recv);
    mlp_head_kernel<<<dim3(BATCH / 16), 256, 0, stream>>>(
        recv, fc1_w, fc1_b, fc2_w, fc2_b, out_w, out_b, (float*)d_out);
}
